// CompositionalLearner_87230785782205
// MI455X (gfx1250) — compile-verified, matrix operand order checked by reading, not run on hardware
//
#include <hip/hip_runtime.h>
#include <hip/hip_bf16.h>

// Problem constants (from reference)
#define B_      32
#define L_      48
#define M_      16
#define V_      512
#define TSTEPS  47      // L-1
#define K_      4
#define VOCAB_  64
#define NTYPES_ 16
#define NCOMP   95      // 48 row-coeffs + 47 bias-coeffs
#define KEXT    96      // GEMM K: 64 vocab + 16 type rows, padded to 3*32
#define NCOLS   8192    // M_*V_
#define NTILES  512     // NCOLS/16
#define KSLABS  3       // KEXT/32

typedef __attribute__((ext_vector_type(16))) _Float16 v16h;
typedef __attribute__((ext_vector_type(8)))  float    v8f;

// WMMA f16 operand K-mapping within a 32-wide slab (ISA 7.12.2, wave32):
// lane half = lane>>4; packed element e -> K = (e<8 ? e : e+8) + 8*half
__device__ __forceinline__ int kof_map(int e, int half) {
    return (e < 8 ? e : e + 8) + half * 8;
}

// ---------------------------------------------------------------------------
// Kernel 1: per-batch affine-coefficient scan.
// Merge weights are data-independent -> the 47-step scan is affine in the
// initial embeddings. Thread c owns coefficient component c (0..47: weight on
// initial row j=c; 48..94: weight on type_bias of step t=c-48); its 48-row
// state lives in its own LDS column (no barriers in the scan loop).
// Epilogue buckets coefficients by vocab id / type id into Cext[B][KEXT].
// ---------------------------------------------------------------------------
__global__ void __launch_bounds__(128)
scan_coef_kernel(const int* __restrict__ ids,        // [B, L]
                 const int* __restrict__ positions,  // [B, 47]
                 const int* __restrict__ types,      // [B, 47]
                 const int* __restrict__ spans,      // [B, 47]
                 const float* __restrict__ w_score,  // [16, 4]
                 _Float16* __restrict__ Cext)        // [B, KEXT]
{
    __shared__ float st[L_][128];   // [row][component]
    const int b   = blockIdx.x;
    const int tid = threadIdx.x;

    if (tid < NCOMP) {
        for (int r = 0; r < L_; ++r)
            st[r][tid] = (tid < L_ && r == tid) ? 1.0f : 0.0f;

        for (int t = 0; t < TSTEPS; ++t) {
            const int p  = positions[b * TSTEPS + t];
            const int s  = spans[b * TSTEPS + t];
            const int ty = types[b * TSTEPS + t];
            // softmax over masked w_score row (GUMBEL_TEMP = 1)
            float lg[K_], mx = -3.0e38f;
            #pragma unroll
            for (int k = 0; k < K_; ++k) {
                lg[k] = (k < s) ? w_score[ty * K_ + k] : -1.0e9f;
                mx = fmaxf(mx, lg[k]);
            }
            float w[K_], sum = 0.0f;
            #pragma unroll
            for (int k = 0; k < K_; ++k) { w[k] = __expf(lg[k] - mx); sum += w[k]; }
            const float inv = 1.0f / sum;
            // merged row coefficients (convex combination of rows p..p+3)
            float merged = 0.0f;
            #pragma unroll
            for (int k = 0; k < K_; ++k) {
                int g = p + k; if (g > L_ - 1) g = L_ - 1;
                merged += (w[k] * inv) * st[g][tid];
            }
            if (tid == 48 + t) merged += 1.0f;   // bias injected at step t
            // ragged shift left by s-1 (ascending j: src >= j, safe in place)
            for (int j = p + 1; j < L_; ++j) {
                int src = j + s - 1;
                st[j][tid] = (src < L_) ? st[src][tid] : 0.0f;
            }
            st[p][tid] = merged;
        }
    }
    __syncthreads();   // epilogue reads other threads' columns of row 0

    if (tid < KEXT) {
        float acc = 0.0f;
        if (tid < VOCAB_) {                       // bucket row-coeffs by vocab id
            for (int j = 0; j < L_; ++j)
                if (ids[b * L_ + j] == tid) acc += st[0][j];
        } else if (tid < VOCAB_ + NTYPES_) {      // bucket bias-coeffs by type
            const int ty = tid - VOCAB_;
            for (int t = 0; t < TSTEPS; ++t)
                if (types[b * TSTEPS + t] == ty) acc += st[0][48 + t];
        }                                          // 80..95: zero K padding
        Cext[b * KEXT + tid] = (_Float16)acc;
    }
}

// ---------------------------------------------------------------------------
// Kernel 2: build B directly in the WMMA register layout (pre-swizzled):
//   Bswz[((kk*NTILES + nt)*32 + lane)*16 + e] = B[K = kk*32 + kof(e,half)]
//                                                [n = nt*16 + (lane&15)]
// where logical B rows are: 0..63 emb_dec[w] flattened over (m,v);
// 64..79 type_bias[ty][v] (bias is m-broadcast); 80..95 zero padding.
// GEMM then reads each lane's operand as one aligned 32-byte vector load.
// ---------------------------------------------------------------------------
__global__ void __launch_bounds__(256)
build_B_kernel(const float* __restrict__ emb_dec,    // [64,16,512]
               const float* __restrict__ type_bias,  // [16,512]
               _Float16* __restrict__ Bswz)          // [KSLABS*NTILES*32*16]
{
    const int idx = blockIdx.x * blockDim.x + threadIdx.x;
    if (idx >= KSLABS * NTILES * 32 * 16) return;
    const int e    = idx & 15;
    const int lane = (idx >> 4) & 31;
    const int nt   = (idx >> 9) & (NTILES - 1);
    const int kk   = idx >> 18;
    const int K    = kk * 32 + kof_map(e, lane >> 4);
    const int n    = nt * 16 + (lane & 15);
    float v = 0.0f;
    if (K < VOCAB_)                   v = emb_dec[K * NCOLS + n];
    else if (K < VOCAB_ + NTYPES_)    v = type_bias[(K - VOCAB_) * V_ + (n & (V_ - 1))];
    Bswz[idx] = (_Float16)v;
}

// ---------------------------------------------------------------------------
// Kernel 2b: swizzle tiny A (Cext[32,96]) into WMMA register layout:
//   Aswz[((kk*2 + mt)*32 + lane)*16 + e] = Cext[mt*16 + (lane&15)]
//                                              [kk*32 + kof(e,half)]
// ---------------------------------------------------------------------------
__global__ void __launch_bounds__(256)
swizzle_A_kernel(const _Float16* __restrict__ Cext,  // [32, KEXT]
                 _Float16* __restrict__ Aswz)        // [KSLABS*2*32*16]
{
    const int idx = blockIdx.x * blockDim.x + threadIdx.x;
    if (idx >= KSLABS * 2 * 32 * 16) return;
    const int e    = idx & 15;
    const int lane = (idx >> 4) & 31;
    const int mt   = (idx >> 9) & 1;
    const int kk   = idx >> 10;
    const int m    = mt * 16 + (lane & 15);
    const int K    = kk * 32 + kof_map(e, lane >> 4);
    Aswz[idx] = Cext[m * KEXT + K];
}

// ---------------------------------------------------------------------------
// Kernel 3: WMMA GEMM  Out[32,8192] = Cext[32,96] @ B[96,8192]  (f32 accum)
// One 16x16 tile per wave32: 2 M-tiles x 512 N-tiles = 1024 waves, 3 k-steps.
// Operands pre-swizzled -> per-lane v16h = one aligned 32B load (2x b128).
// D layout: lane = N col, vgpr r -> M = r + 8*(lane>=16).
// ---------------------------------------------------------------------------
__global__ void __launch_bounds__(128)
wmma_gemm_kernel(const _Float16* __restrict__ Aswz,  // pre-swizzled A
                 const _Float16* __restrict__ Bswz,  // pre-swizzled B
                 float* __restrict__ Out)            // [32, 8192]
{
    const int wave = blockIdx.x * (blockDim.x >> 5) + (threadIdx.x >> 5);
    const int lane = threadIdx.x & 31;
    const int mt   = wave >> 9;     // 0..1
    const int nt   = wave & 511;    // 0..511
    const int half = lane >> 4;
    const int mn   = lane & 15;

    // prefetch next k-slab's B block (lowers to global_prefetch_b8)
    __builtin_prefetch(Bswz + (((size_t)(1 * NTILES + nt) * 32 + lane) << 4), 0, 1);

    v8f acc = {};
    for (int kk = 0; kk < KSLABS; ++kk) {
        const v16h a  = *(const v16h*)(Aswz + (((size_t)(kk * 2 + mt) * 32 + lane) << 4));
        const v16h bv = *(const v16h*)(Bswz + (((size_t)(kk * NTILES + nt) * 32 + lane) << 4));
        // (neg_a, A, neg_b, B, c_mod, C, reuse_a, reuse_b)
        acc = __builtin_amdgcn_wmma_f32_16x16x32_f16(
                  false, a, false, bv, (short)0, acc, false, false);
    }
    #pragma unroll
    for (int r = 0; r < 8; ++r) {
        const int m = mt * 16 + r + half * 8;
        const int n = nt * 16 + mn;
        Out[(size_t)m * NCOLS + n] = acc[r];
    }
}

// ---------------------------------------------------------------------------
// Kernel 4: softmax over V=512 per (b,m) row. Gout flat layout (b*8192+m*512+v)
// equals output layout ((b*16+m)*512+v), so rows map 1:1.
// ---------------------------------------------------------------------------
__global__ void __launch_bounds__(256)
softmax_kernel(const float* __restrict__ Gout, float* __restrict__ out)
{
    const int row = blockIdx.x;        // 0..511
    const int tid = threadIdx.x;       // 256 threads, 2 elems each
    __shared__ float red[256];
    const float* src = Gout + (size_t)row * V_;
    const float x0 = src[tid], x1 = src[tid + 256];

    red[tid] = fmaxf(x0, x1);
    __syncthreads();
    for (int s = 128; s > 0; s >>= 1) {
        if (tid < s) red[tid] = fmaxf(red[tid], red[tid + s]);
        __syncthreads();
    }
    const float mx = red[0];
    __syncthreads();
    const float e0 = __expf(x0 - mx), e1 = __expf(x1 - mx);
    red[tid] = e0 + e1;
    __syncthreads();
    for (int s = 128; s > 0; s >>= 1) {
        if (tid < s) red[tid] += red[tid + s];
        __syncthreads();
    }
    const float inv = 1.0f / red[0];
    out[(size_t)row * V_ + tid]       = e0 * inv;
    out[(size_t)row * V_ + tid + 256] = e1 * inv;
}

// ---------------------------------------------------------------------------
extern "C" void kernel_launch(void* const* d_in, const int* in_sizes, int n_in,
                              void* d_out, int out_size, void* d_ws, size_t ws_size,
                              hipStream_t stream) {
    const int*   ids       = (const int*)d_in[0];
    const int*   positions = (const int*)d_in[1];
    const int*   types     = (const int*)d_in[2];
    const int*   spans     = (const int*)d_in[3];
    const float* emb_dec   = (const float*)d_in[4];
    // d_in[5] = emb_term: provably dead (output reads only dec[:,0])
    const float* w_score   = (const float*)d_in[6];
    const float* type_bias = (const float*)d_in[7];
    float* out = (float*)d_out;

    char* ws = (char*)d_ws;
    _Float16* Cext = (_Float16*)ws;                      //   6,144 B
    _Float16* Aswz = (_Float16*)(ws + 8192);             //   6,144 B (32B aligned)
    _Float16* Bswz = (_Float16*)(ws + 16384);            // 1,572,864 B (32B aligned)
    float*    Gout = (float*)(ws + 16384 + (size_t)KSLABS * NTILES * 32 * 16 * 2);

    scan_coef_kernel<<<B_, 128, 0, stream>>>(ids, positions, types, spans,
                                             w_score, Cext);
    build_B_kernel<<<(KSLABS * NTILES * 32 * 16 + 255) / 256, 256, 0, stream>>>(
        emb_dec, type_bias, Bswz);
    swizzle_A_kernel<<<(KSLABS * 2 * 32 * 16 + 255) / 256, 256, 0, stream>>>(Cext, Aswz);
    wmma_gemm_kernel<<<256, 128, 0, stream>>>(Aswz, Bswz, Gout);  // 1024 waves
    softmax_kernel<<<B_ * M_, 256, 0, stream>>>(Gout, out);
}
